// Actions_67783173865693
// MI455X (gfx1250) — matrix operand order repaired and checked by reading, not compiled, not run on hardware
//
#include <hip/hip_runtime.h>
#include <hip/hip_bf16.h>

typedef __attribute__((ext_vector_type(16))) _Float16 v16h;
typedef __attribute__((ext_vector_type(8)))  _Float16 v8h;
typedef __attribute__((ext_vector_type(8)))  float    v8f;

// ---------------- packed-weight layout offsets (in halves) ----------------
#define OFF_EW0  0LL
#define OFF_EW1  (OFF_EW0 + 384LL*256)   //  98304
#define OFF_EW2  (OFF_EW1 + 256LL*256)   // 163840
#define OFF_EW3  (OFF_EW2 + 256LL*256)   // 229376
#define OFF_TW0  (OFF_EW3 + 256LL*16)    // 233472
#define OFF_TW1  (OFF_TW0 + 608LL*256)   // 389120
#define OFF_TW2  (OFF_TW1 + 256LL*256)
#define OFF_TW3  (OFF_TW2 + 256LL*256)
#define PACK_TOTAL_HALVES (OFF_TW3 + 256LL*16)   // 524288 halves = 1 MiB

__device__ __forceinline__ float silu_f(float x) {
    return x / (1.0f + __expf(-x));
}

// ---- A fragment from LDS (contiguous k order, same as B):
// lanes 0-15 : M=lane, halves 0-15 = K kb+0..15
// lanes 16-31: M=lane-16, halves 0-15 = K kb+16..31
__device__ __forceinline__ v16h load_a_frag(const _Float16* x, int ld, int kb, int lane) {
    const int m  = lane & 15;
    const int hi = lane >> 4;
    const _Float16* p = x + m * ld + kb + hi * 16;
    v8h a0 = *(const v8h*)(p);        // ds_load_b128
    v8h a1 = *(const v8h*)(p + 8);    // ds_load_b128
    v16h a;
#pragma unroll
    for (int i = 0; i < 8; ++i) { a[i] = a0[i]; a[8 + i] = a1[i]; }
    return a;
}

// ---- B fragment, prepacked f16 (2x global_load_b128 per fragment)
__device__ __forceinline__ v16h load_b_packed(const _Float16* __restrict__ P,
                                              int kf, int nt, int kfrags, int lane) {
    const _Float16* p = P + ((((long long)nt * kfrags + kf) * 32) + lane) * 16;
    v8h b0 = *(const v8h*)(p);
    v8h b1 = *(const v8h*)(p + 8);
    v16h b;
#pragma unroll
    for (int i = 0; i < 8; ++i) { b[i] = b0[i]; b[8 + i] = b1[i]; }
    return b;
}

// ---- B fragment directly from f32 weights (fallback if ws too small)
// B 32x16 frag (contiguous k, as A): lane holds column N = n0 + lane%16; halves = K kb + (lane/16)*16 + i
__device__ __forceinline__ v16h load_b_direct(const float* __restrict__ W,
                                              int Kact, int Nact, int kb, int n0, int lane) {
    const int n  = n0 + (lane & 15);
    const int k0 = kb + (lane >> 4) * 16;
    v16h b;
#pragma unroll
    for (int i = 0; i < 16; ++i) {
        int k = k0 + i;
        float w = (k < Kact && n < Nact) ? W[(long long)k * Nact + n] : 0.0f;
        b[i] = (_Float16)w;
    }
    return b;
}

// ---- one MLP layer on a 32-row tile (two 16-row M-tiles share every B frag)
template<bool PACKED, bool FINAL>
__device__ __forceinline__ void mlp_layer2(const _Float16* xin, int ldin, int Kpad, int Kact,
                                           const float* __restrict__ W,
                                           const _Float16* __restrict__ Wp,
                                           int Nact, _Float16* xout,
                                           float* __restrict__ gout, long long row0,
                                           long long nrows, const float* __restrict__ bias,
                                           int lane) {
    const int kfrags = Kpad >> 5;
    const int ntiles = FINAL ? 1 : 16;
    const int n  = lane & 15;
    const int hi = lane >> 4;
    for (int nt = 0; nt < ntiles; ++nt) {
        v8f acc0 = {0.f, 0.f, 0.f, 0.f, 0.f, 0.f, 0.f, 0.f};
        v8f acc1 = {0.f, 0.f, 0.f, 0.f, 0.f, 0.f, 0.f, 0.f};
        for (int kf = 0; kf < kfrags; ++kf) {
            v16h b  = PACKED ? load_b_packed(Wp, kf, nt, kfrags, lane)
                             : load_b_direct(W, Kact, Nact, kf * 32, nt * 16, lane);
            v16h a0 = load_a_frag(xin,             ldin, kf * 32, lane);
            v16h a1 = load_a_frag(xin + 16 * ldin, ldin, kf * 32, lane);
            acc0 = __builtin_amdgcn_wmma_f32_16x16x32_f16(
                false, a0, false, b, (short)0, acc0, false, false);
            acc1 = __builtin_amdgcn_wmma_f32_16x16x32_f16(
                false, a1, false, b, (short)0, acc1, false, false);
        }
        if (FINAL) {
            if (n < Nact) {
                float bv = bias[n];
#pragma unroll
                for (int r = 0; r < 8; ++r) {
                    long long m0 = row0 + hi * 8 + r;
                    long long m1 = row0 + 16 + hi * 8 + r;
                    if (m0 < nrows) gout[m0 * Nact + n] = acc0[r] + bv;
                    if (m1 < nrows) gout[m1 * Nact + n] = acc1[r] + bv;
                }
            }
        } else {
#pragma unroll
            for (int r = 0; r < 8; ++r) {
                int m = hi * 8 + r;
                xout[m * 256 + nt * 16 + n]        = (_Float16)silu_f(acc0[r]);
                xout[(16 + m) * 256 + nt * 16 + n] = (_Float16)silu_f(acc1[r]);
            }
        }
    }
}

__device__ __forceinline__ void store_h_row_f16(const float* __restrict__ hrow,
                                                _Float16* __restrict__ q, int lane) {
    const float4 hv = ((const float4*)hrow)[lane];     // global_load_b128
    _Float16* p = q + lane * 4;
    p[0] = (_Float16)hv.x; p[1] = (_Float16)hv.y;
    p[2] = (_Float16)hv.z; p[3] = (_Float16)hv.w;      // ds_store_b64
}

// ---------------- edge MLP: [E] x (356 -> 256 -> 256 -> 256 -> 3) ----------------
template<bool PACKED>
__global__ void __launch_bounds__(32)
edge_mlp_kernel(const float* __restrict__ h, const int* __restrict__ src,
                const int* __restrict__ dst, const float* __restrict__ enorm,
                const float* __restrict__ eW0, const float* __restrict__ eW1,
                const float* __restrict__ eW2, const float* __restrict__ eW3,
                const float* __restrict__ eb3, const _Float16* __restrict__ wp,
                float* __restrict__ out, long long E) {
    __shared__ __align__(16) _Float16 bufA[32 * 384];
    __shared__ __align__(16) _Float16 bufB[32 * 256];
    const int lane = threadIdx.x;
    const long long row0 = (long long)blockIdx.x * 32;

    // stage e_in = [h[src] | h[dst] | RBF(enorm)] as f16, zero-padded to 384
    for (int r = 0; r < 32; ++r) {
        long long e = row0 + r; if (e >= E) e = E - 1;
        const int s = src[e], d = dst[e];
        const float rn = enorm[e];
        store_h_row_f16(h + (long long)s * 128, bufA + r * 384, lane);
        store_h_row_f16(h + (long long)d * 128, bufA + r * 384 + 128, lane);
#pragma unroll
        for (int j = 0; j < 4; ++j) {
            int c = 256 + lane * 4 + j;       // covers cols 256..383
            float v = 0.0f;
            if (c < 356) { float dd = 0.1f * (c - 256) - rn; v = __expf(-10.0f * dd * dd); }
            bufA[r * 384 + c] = (_Float16)v;
        }
    }
    __syncthreads();
    mlp_layer2<PACKED, false>(bufA, 384, 384, 356, eW0, wp + OFF_EW0, 256, bufB, nullptr, 0, 0, nullptr, lane);
    __syncthreads();
    mlp_layer2<PACKED, false>(bufB, 256, 256, 256, eW1, wp + OFF_EW1, 256, bufA, nullptr, 0, 0, nullptr, lane);
    __syncthreads();
    mlp_layer2<PACKED, false>(bufA, 256, 256, 256, eW2, wp + OFF_EW2, 256, bufB, nullptr, 0, 0, nullptr, lane);
    __syncthreads();
    mlp_layer2<PACKED, true >(bufB, 256, 256, 256, eW3, wp + OFF_EW3, 3, nullptr, out, row0, E, eb3, lane);
}

// ---------------- triplet MLP: [T] x (586 -> 256 -> 256 -> 256 -> 3) ----------------
template<bool PACKED>
__global__ void __launch_bounds__(32)
trip_mlp_kernel(const float* __restrict__ h, const int* __restrict__ ts,
                const int* __restrict__ ti, const int* __restrict__ tj,
                const float* __restrict__ nij, const float* __restrict__ nik,
                const float* __restrict__ cosv, const float* __restrict__ sinv,
                const float* __restrict__ tW0, const float* __restrict__ tW1,
                const float* __restrict__ tW2, const float* __restrict__ tW3,
                const float* __restrict__ tb3, const _Float16* __restrict__ wp,
                float* __restrict__ out, long long T) {
    __shared__ __align__(16) _Float16 bufA[32 * 608];
    __shared__ __align__(16) _Float16 bufB[32 * 256];
    const int lane = threadIdx.x;
    const long long row0 = (long long)blockIdx.x * 32;

    for (int r = 0; r < 32; ++r) {
        long long e = row0 + r; if (e >= T) e = T - 1;
        const int s = ts[e], i1 = ti[e], j1 = tj[e];
        const float rij = nij[e], rik = nik[e], cs = cosv[e], sn = sinv[e];
        store_h_row_f16(h + (long long)s  * 128, bufA + r * 608, lane);
        store_h_row_f16(h + (long long)i1 * 128, bufA + r * 608 + 128, lane);
        store_h_row_f16(h + (long long)j1 * 128, bufA + r * 608 + 256, lane);
        for (int c = 384 + lane; c < 608; c += 32) {   // 7 iterations
            float v;
            if (c < 484)       { float dd = 0.1f * (c - 384) - rij; v = __expf(-10.0f * dd * dd); }
            else if (c < 584)  { float dd = 0.1f * (c - 484) - rik; v = __expf(-10.0f * dd * dd); }
            else if (c == 584) v = cs;
            else if (c == 585) v = sn;
            else               v = 0.0f;
            bufA[r * 608 + c] = (_Float16)v;
        }
    }
    __syncthreads();
    mlp_layer2<PACKED, false>(bufA, 608, 608, 586, tW0, wp + OFF_TW0, 256, bufB, nullptr, 0, 0, nullptr, lane);
    __syncthreads();
    mlp_layer2<PACKED, false>(bufB, 256, 256, 256, tW1, wp + OFF_TW1, 256, bufA, nullptr, 0, 0, nullptr, lane);
    __syncthreads();
    mlp_layer2<PACKED, false>(bufA, 256, 256, 256, tW2, wp + OFF_TW2, 256, bufB, nullptr, 0, 0, nullptr, lane);
    __syncthreads();
    mlp_layer2<PACKED, true >(bufB, 256, 256, 256, tW3, wp + OFF_TW3, 3, nullptr, out, row0, T, tb3, lane);
}

// ---------------- pack f32 weights into WMMA-B f16 register layout ----------------
__global__ void __launch_bounds__(256)
pack_weights_kernel(const float* __restrict__ W, _Float16* __restrict__ dstp,
                    int Kact, int Nact, int Kpad, int Npad) {
    const int kfrags = Kpad >> 5;
    const long long total = (long long)kfrags * (Npad >> 4) * 512;
    long long idx = (long long)blockIdx.x * blockDim.x + threadIdx.x;
    if (idx >= total) return;
    const int i    = (int)(idx & 15);
    const int lane = (int)((idx >> 4) & 31);
    const long long frag = idx >> 9;
    const int kf = (int)(frag % kfrags);
    const int nt = (int)(frag / kfrags);
    const int n = nt * 16 + (lane & 15);
    const int k = kf * 32 + (lane >> 4) * 16 + i;
    float w = (k < Kact && n < Nact) ? W[(long long)k * Nact + n] : 0.0f;
    dstp[idx] = (_Float16)w;
}

extern "C" void kernel_launch(void* const* d_in, const int* in_sizes, int n_in,
                              void* d_out, int out_size, void* d_ws, size_t ws_size,
                              hipStream_t stream) {
    const float* h     = (const float*)d_in[0];
    const int*   src   = (const int*)d_in[1];
    const int*   dst   = (const int*)d_in[2];
    const float* enorm = (const float*)d_in[3];
    const int*   ts    = (const int*)d_in[4];
    const int*   ti    = (const int*)d_in[5];
    const int*   tj    = (const int*)d_in[6];
    const float* nij   = (const float*)d_in[7];
    const float* nik   = (const float*)d_in[8];
    const float* cosv  = (const float*)d_in[9];
    const float* sinv  = (const float*)d_in[10];
    // d_in[11] = mu (recomputed on device as 0.1*j)
    const float* eW0 = (const float*)d_in[12];
    const float* eW1 = (const float*)d_in[13];
    const float* eW2 = (const float*)d_in[14];
    const float* eW3 = (const float*)d_in[15];
    const float* eb3 = (const float*)d_in[16];
    const float* tW0 = (const float*)d_in[17];
    const float* tW1 = (const float*)d_in[18];
    const float* tW2 = (const float*)d_in[19];
    const float* tW3 = (const float*)d_in[20];
    const float* tb3 = (const float*)d_in[21];

    const long long E = in_sizes[1];
    const long long T = in_sizes[4];
    float* out_e = (float*)d_out;
    float* out_t = (float*)d_out + E * 3;

    const bool packed = ws_size >= (size_t)(PACK_TOTAL_HALVES * 2);
    _Float16* wp = (_Float16*)d_ws;

    const int eblocks = (int)((E + 31) / 32);
    const int tblocks = (int)((T + 31) / 32);

    if (packed) {
        struct { const float* W; long long off; int Ka, Na, Kp, Np; } jobs[8] = {
            {eW0, OFF_EW0, 356, 256, 384, 256}, {eW1, OFF_EW1, 256, 256, 256, 256},
            {eW2, OFF_EW2, 256, 256, 256, 256}, {eW3, OFF_EW3, 256, 3,   256, 16 },
            {tW0, OFF_TW0, 586, 256, 608, 256}, {tW1, OFF_TW1, 256, 256, 256, 256},
            {tW2, OFF_TW2, 256, 256, 256, 256}, {tW3, OFF_TW3, 256, 3,   256, 16 },
        };
        for (int j = 0; j < 8; ++j) {
            long long total = (long long)(jobs[j].Kp >> 5) * (jobs[j].Np >> 4) * 512;
            int blocks = (int)((total + 255) / 256);
            pack_weights_kernel<<<blocks, 256, 0, stream>>>(
                jobs[j].W, wp + jobs[j].off, jobs[j].Ka, jobs[j].Na, jobs[j].Kp, jobs[j].Np);
        }
        edge_mlp_kernel<true><<<eblocks, 32, 0, stream>>>(
            h, src, dst, enorm, eW0, eW1, eW2, eW3, eb3, wp, out_e, E);
        trip_mlp_kernel<true><<<tblocks, 32, 0, stream>>>(
            h, ts, ti, tj, nij, nik, cosv, sinv, tW0, tW1, tW2, tW3, tb3, wp, out_t, T);
    } else {
        edge_mlp_kernel<false><<<eblocks, 32, 0, stream>>>(
            h, src, dst, enorm, eW0, eW1, eW2, eW3, eb3, wp, out_e, E);
        trip_mlp_kernel<false><<<tblocks, 32, 0, stream>>>(
            h, ts, ti, tj, nij, nik, cosv, sinv, tW0, tW1, tW2, tW3, tb3, wp, out_t, T);
    }
}
